// TransformerBlock_25374666785270
// MI455X (gfx1250) — matrix operand order repaired and checked by reading, not compiled, not run on hardware
//
#include <hip/hip_runtime.h>
#include <math.h>

// ---------------- problem constants ----------------
static constexpr int B_  = 4;
static constexpr int N_  = 2048;
static constexpr int D_  = 1024;
static constexpr int H_  = 16;
static constexpr int HD_ = 64;
static constexpr int L_  = 4;
static constexpr int R_  = 16;
static constexpr int KT_ = 204;   // ktop = int(2048 * 0.1)
static constexpr int KP_ = 208;   // padded to multiple of 16
static constexpr int D4_ = 4096;

typedef __attribute__((ext_vector_type(16))) __bf16 bf16x16;
typedef __attribute__((ext_vector_type(8)))  __bf16 bf16x8;
typedef __attribute__((ext_vector_type(8)))  float  f32x8;
typedef __attribute__((ext_vector_type(4)))  int    v4i;

// CDNA5 async global->LDS staging (ASYNCcnt-tracked), with safe fallback.
#if defined(__AMDGCN__) && __has_builtin(__builtin_amdgcn_global_load_async_to_lds_b128) && \
    __has_builtin(__builtin_amdgcn_s_wait_asynccnt)
#define USE_ASYNC_LDS 1
#else
#define USE_ASYNC_LDS 0
#endif

// =====================================================================
// Generic bf16 WMMA GEMM: C[M,N] = act(A[M,K] @ Bw[K,N] + bias) (+ C)
// block = 256 threads = 8 waves; block tile 128x128; k-step 64.
// Each wave computes a 32x64 sub-tile = 2x4 of 16x16 WMMA tiles,
// 16 v_wmma_f32_16x16x32_bf16 per wave per barrier pair.
// =====================================================================
template <int ACT_GELU, int OUT_BF16, int ACCUM>
__global__ void __launch_bounds__(256)
gemm_wmma(const __bf16* __restrict__ A, const __bf16* __restrict__ Bw,
          const float* __restrict__ bias, float* __restrict__ Cf,
          __bf16* __restrict__ Cb, int M, int N, int K)
{
    __shared__ __bf16 As[128][64];   // [m][k]
    __shared__ __bf16 Bs[128][64];   // [n][k]  (transposed tile)

    const int t    = threadIdx.x;
    const int lane = t & 31;
    const int wave = t >> 5;
    const int wr   = wave & 3;       // 4 row groups of 32
    const int wc   = wave >> 2;      // 2 col groups of 64
    const int r    = lane & 15;
    const int kh   = lane >> 4;      // k-half select (wave32 frag layout)

    const int bm = blockIdx.y * 128;
    const int bn = blockIdx.x * 128;

    f32x8 acc[2][4];
#pragma unroll
    for (int i = 0; i < 2; ++i)
#pragma unroll
        for (int j = 0; j < 4; ++j)
#pragma unroll
            for (int e = 0; e < 8; ++e) acc[i][j][e] = 0.0f;

    // A staging: each thread copies 64B of one row (row t>>1, col half (t&1)*32)
    const int arow = t >> 1;
    const int acol = (t & 1) << 5;   // 0 or 32
    // B staging: each thread handles an 8(K) x 4(N) patch
    const int bk0 = (t >> 5) << 3;   // k0 = (t/32)*8
    const int bnn = (t & 31) << 2;   // n  = (t%32)*4
    long aRowG = (long)bm + arow;
    if (aRowG > (long)M - 1) aRowG = (long)M - 1;   // clamp: reads stay in-bounds

    for (int kt = 0; kt < K; kt += 64) {
        // ---- stage A tile ----
        {
            const __bf16* ga = A + aRowG * (long)K + kt + acol;
#if USE_ASYNC_LDS
#pragma unroll
            for (int c = 0; c < 4; ++c)
                __builtin_amdgcn_global_load_async_to_lds_b128(
                    (__attribute__((address_space(1))) v4i*)(ga + c * 8),
                    (__attribute__((address_space(3))) v4i*)&As[arow][acol + c * 8],
                    0, 0);
#else
            const uint4* sa = reinterpret_cast<const uint4*>(ga);
            uint4 a0 = sa[0], a1 = sa[1], a2 = sa[2], a3 = sa[3];
            *reinterpret_cast<uint4*>(&As[arow][acol])      = a0;
            *reinterpret_cast<uint4*>(&As[arow][acol + 8])  = a1;
            *reinterpret_cast<uint4*>(&As[arow][acol + 16]) = a2;
            *reinterpret_cast<uint4*>(&As[arow][acol + 24]) = a3;
#endif
        }
        // ---- stage B tile transposed into Bs[n][k]: 8 coalesced 8B loads,
        //      then one ds_store_b128 per output column (k-contiguous) ----
        {
            const __bf16* sb = Bw + (long)(kt + bk0) * N + bn + bnn;
            __bf16 tmp[8][4];
#pragma unroll
            for (int dk = 0; dk < 8; ++dk)
                *reinterpret_cast<uint2*>(&tmp[dk][0]) =
                    *reinterpret_cast<const uint2*>(sb + (long)dk * N);
#pragma unroll
            for (int dn = 0; dn < 4; ++dn) {
                __bf16 col[8];
#pragma unroll
                for (int dk = 0; dk < 8; ++dk) col[dk] = tmp[dk][dn];
                *reinterpret_cast<uint4*>(&Bs[bnn + dn][bk0]) =
                    *reinterpret_cast<uint4*>(col);
            }
        }
        if (kt + 64 < K) {  // emits global_prefetch_b8
            __builtin_prefetch(A + aRowG * (long)K + kt + 64 + acol, 0, 1);
            __builtin_prefetch(Bw + (long)(kt + 64 + bk0) * N + bn + bnn, 0, 1);
        }
#if USE_ASYNC_LDS
        __builtin_amdgcn_s_wait_asynccnt(0);
#endif
        __syncthreads();

        // ---- two k-slices of 32 per staged tile ----
#pragma unroll
        for (int ks = 0; ks < 2; ++ks) {
            bf16x16 afr[2];
#pragma unroll
            for (int it = 0; it < 2; ++it) {
                const __bf16* base = &As[wr * 32 + it * 16 + r][ks * 32];
                bf16x8 lo = *reinterpret_cast<const bf16x8*>(base + kh * 16);
                bf16x8 hi = *reinterpret_cast<const bf16x8*>(base + kh * 16 + 8);
                bf16x16 av;
#pragma unroll
                for (int e = 0; e < 8; ++e) { av[e] = lo[e]; av[8 + e] = hi[e]; }
                afr[it] = av;
            }
            bf16x16 bfr[4];
#pragma unroll
            for (int jt = 0; jt < 4; ++jt)
                bfr[jt] = *reinterpret_cast<const bf16x16*>(
                    &Bs[wc * 64 + jt * 16 + r][ks * 32 + kh * 16]);

#pragma unroll
            for (int it = 0; it < 2; ++it)
#pragma unroll
                for (int jt = 0; jt < 4; ++jt)
                    acc[it][jt] = __builtin_amdgcn_wmma_f32_16x16x32_bf16(
                        false, afr[it], false, bfr[jt], (short)0, acc[it][jt], false, false);
        }
        __syncthreads();
    }

    // ---- epilogue (C layout: lanes 0-15 -> M=vgpr, lanes 16-31 -> M=8+vgpr) ----
    const int nloc  = lane & 15;
    const int mhalf = (lane >> 4) * 8;
#pragma unroll
    for (int it = 0; it < 2; ++it) {
#pragma unroll
        for (int jt = 0; jt < 4; ++jt) {
            const int col = bn + wc * 64 + jt * 16 + nloc;
            const float bval = bias ? bias[col] : 0.0f;
#pragma unroll
            for (int e = 0; e < 8; ++e) {
                const int row = bm + wr * 32 + it * 16 + mhalf + e;
                if (row < M) {
                    float v = acc[it][jt][e] + bval;
                    if (ACT_GELU) v = 0.5f * v * (1.0f + erff(v * 0.70710678118654752f));
                    const long off = (long)row * N + col;
                    if (ACCUM) v += Cf[off];
                    if (OUT_BF16) Cb[off] = (__bf16)v;
                    else          Cf[off] = v;
                }
            }
        }
    }
}

// =====================================================================
// Router
// =====================================================================
__global__ void colmean_kernel(const float* __restrict__ x, float* __restrict__ xbar)
{
    const int tid = blockIdx.x * blockDim.x + threadIdx.x;
    if (tid >= B_ * D_) return;
    const int b = tid / D_, d = tid % D_;
    const float* p = x + (long)b * N_ * D_ + d;
    float s = 0.f;
    for (int n = 0; n < N_; ++n) s += p[(long)n * D_];
    xbar[tid] = s * (1.0f / N_);
}

__global__ void qglobal_kernel(const float* __restrict__ xbar,
                               const float* __restrict__ wq_r, float* __restrict__ qg)
{
    const int t = threadIdx.x;
    if (t >= B_ * R_) return;
    const int b = t / R_, rr = t % R_;
    float s = 0.f;
    for (int d = 0; d < D_; ++d) s += xbar[b * D_ + d] * wq_r[d * R_ + rr];
    qg[t] = s;
}

__global__ void wvec_kernel(const float* __restrict__ wk_r,
                            const float* __restrict__ qg, float* __restrict__ wv)
{
    const int tid = blockIdx.x * blockDim.x + threadIdx.x;
    if (tid >= B_ * D_) return;
    const int b = tid / D_, d = tid % D_;
    float s = 0.f;
#pragma unroll
    for (int rr = 0; rr < R_; ++rr) s += wk_r[d * R_ + rr] * qg[b * R_ + rr];
    wv[tid] = s;
}

__global__ void __launch_bounds__(256)
scores_kernel(const float* __restrict__ x, const float* __restrict__ wv, float* __restrict__ sc)
{
    __shared__ float red[256];
    const int row = blockIdx.x;          // b*N + n
    const int b = row / N_;
    const int t = threadIdx.x;
    const float* xr = x + (long)row * D_;
    const float* wb = wv + b * D_;
    float a = 0.f;
#pragma unroll
    for (int i = 0; i < 4; ++i) { const int d = t + i * 256; a += xr[d] * wb[d]; }
    red[t] = a; __syncthreads();
    for (int off = 128; off; off >>= 1) {
        if (t < off) red[t] += red[t + off];
        __syncthreads();
    }
    if (t == 0) sc[row] = red[0];
}

__global__ void __launch_bounds__(256)
topk_kernel(const float* __restrict__ sc, int* __restrict__ idx, int* __restrict__ pos)
{
    __shared__ float s[N_];
    __shared__ float rv[256];
    __shared__ int   ri[256];
    const int b = blockIdx.x, t = threadIdx.x;
    for (int n = t; n < N_; n += 256) { s[n] = sc[b * N_ + n]; pos[b * N_ + n] = -1; }
    __syncthreads();
    for (int j = 0; j < KT_; ++j) {
        float best = -3.4e38f; int bi = 0;
        for (int n = t; n < N_; n += 256) { float v = s[n]; if (v > best) { best = v; bi = n; } }
        rv[t] = best; ri[t] = bi; __syncthreads();
        for (int off = 128; off; off >>= 1) {
            if (t < off && rv[t + off] > rv[t]) { rv[t] = rv[t + off]; ri[t] = ri[t + off]; }
            __syncthreads();
        }
        if (t == 0) { int w = ri[0]; idx[b * KP_ + j] = w; pos[b * N_ + w] = j; s[w] = -3.4e38f; }
        __syncthreads();
    }
}

// =====================================================================
// LayerNorm kernels
// =====================================================================
// LN1: nx = LN(x); gather routed rows (bf16 + f32 copy); x += nx in place.
__global__ void __launch_bounds__(256)
ln1_kernel(float* __restrict__ x_acc, const float* __restrict__ g,
           const float* __restrict__ be, const int* __restrict__ pos,
           __bf16* __restrict__ xs, float* __restrict__ nxr)
{
    __shared__ float s1[256];
    __shared__ float s2[256];
    const int row = blockIdx.x;          // b*N + n
    const int b = row / N_;
    const int t = threadIdx.x;
    float* xr = x_acc + (long)row * D_;
    float v[4]; float a = 0.f, q = 0.f;
#pragma unroll
    for (int i = 0; i < 4; ++i) { v[i] = xr[t + i * 256]; a += v[i]; q += v[i] * v[i]; }
    s1[t] = a; s2[t] = q; __syncthreads();
    for (int off = 128; off; off >>= 1) {
        if (t < off) { s1[t] += s1[t + off]; s2[t] += s2[t + off]; }
        __syncthreads();
    }
    const float mean = s1[0] * (1.0f / D_);
    const float var  = s2[0] * (1.0f / D_) - mean * mean;
    const float rs   = rsqrtf(var + 1e-5f);
    const int j = pos[row];
#pragma unroll
    for (int i = 0; i < 4; ++i) {
        const int d = t + i * 256;
        const float nx = (v[i] - mean) * rs * g[d] + be[d];
        if (j >= 0) {
            const long o = ((long)b * KP_ + j) * D_ + d;
            xs[o]  = (__bf16)nx;
            nxr[o] = nx;
        }
        xr[d] = v[i] + nx;
    }
}

// LN2: n2 = LN(x) -> bf16 full matrix (FFN input)
__global__ void __launch_bounds__(256)
ln2_kernel(const float* __restrict__ x_acc, const float* __restrict__ g,
           const float* __restrict__ be, __bf16* __restrict__ n2)
{
    __shared__ float s1[256];
    __shared__ float s2[256];
    const int row = blockIdx.x;
    const int t = threadIdx.x;
    const float* xr = x_acc + (long)row * D_;
    float v[4]; float a = 0.f, q = 0.f;
#pragma unroll
    for (int i = 0; i < 4; ++i) { v[i] = xr[t + i * 256]; a += v[i]; q += v[i] * v[i]; }
    s1[t] = a; s2[t] = q; __syncthreads();
    for (int off = 128; off; off >>= 1) {
        if (t < off) { s1[t] += s1[t + off]; s2[t] += s2[t + off]; }
        __syncthreads();
    }
    const float mean = s1[0] * (1.0f / D_);
    const float var  = s2[0] * (1.0f / D_) - mean * mean;
    const float rs   = rsqrtf(var + 1e-5f);
#pragma unroll
    for (int i = 0; i < 4; ++i) {
        const int d = t + i * 256;
        n2[(long)row * D_ + d] = (__bf16)((v[i] - mean) * rs * g[d] + be[d]);
    }
}

// =====================================================================
// Attention core (204 tokens per batch, 16 heads of 64) — fp32 VALU.
// One block per (b,h); 8 waves each own a subset of query rows.
// =====================================================================
__global__ void __launch_bounds__(256)
attn_kernel(const float* __restrict__ q, const float* __restrict__ k,
            const float* __restrict__ v, __bf16* __restrict__ o)
{
    __shared__ float pbuf[8][KP_];
    const int bh = blockIdx.x;
    const int b = bh / H_, h = bh % H_;
    const long base = ((long)b * KP_) * D_ + h * HD_;
    const int wave = threadIdx.x >> 5, lane = threadIdx.x & 31;

    for (int jq = wave; jq < KT_; jq += 8) {
        const float* qr = q + base + (long)jq * D_;
        float sc[7];
#pragma unroll
        for (int ii = 0; ii < 7; ++ii) {
            const int jk = lane + ii * 32;
            float s = -3.4e38f;
            if (jk < KT_) {
                const float* kr = k + base + (long)jk * D_;
                float acc = 0.f;
#pragma unroll
                for (int d = 0; d < HD_; ++d) acc += qr[d] * kr[d];
                s = acc * 0.125f;   // 1/sqrt(64)
            }
            sc[ii] = s;
        }
        float mx = -3.4e38f;
#pragma unroll
        for (int ii = 0; ii < 7; ++ii) mx = fmaxf(mx, sc[ii]);
#pragma unroll
        for (int off = 16; off; off >>= 1) mx = fmaxf(mx, __shfl_xor(mx, off));
        float sum = 0.f;
#pragma unroll
        for (int ii = 0; ii < 7; ++ii) {
            const int jk = lane + ii * 32;
            if (jk < KT_) { const float e = __expf(sc[ii] - mx); pbuf[wave][jk] = e; sum += e; }
        }
#pragma unroll
        for (int off = 16; off; off >>= 1) sum += __shfl_xor(sum, off);
        const float inv = 1.0f / sum;
        float a0 = 0.f, a1 = 0.f;
        for (int jk = 0; jk < KT_; ++jk) {
            const float p = pbuf[wave][jk];   // same-wave LDS: in-order, no barrier needed
            const float* vr = v + base + (long)jk * D_;
            a0 += p * vr[lane];
            a1 += p * vr[lane + 32];
        }
        __bf16* orow = o + base + (long)jq * D_;
        orow[lane]      = (__bf16)(a0 * inv);
        orow[lane + 32] = (__bf16)(a1 * inv);
    }
}

// =====================================================================
// Scatter correction: x[b, idx[j]] += o_proj[j] - nx[b, idx[j]]
// =====================================================================
__global__ void __launch_bounds__(256)
scatter_kernel(float* __restrict__ x_acc, const float* __restrict__ oproj,
               const float* __restrict__ nxr, const int* __restrict__ idx)
{
    const int bj = blockIdx.x;           // b*KT + j
    const int b = bj / KT_, j = bj % KT_;
    const int n = idx[b * KP_ + j];
    const long xo = ((long)b * N_ + n) * D_;
    const long so = ((long)b * KP_ + j) * D_;
    const int t = threadIdx.x;
#pragma unroll
    for (int i = 0; i < 4; ++i) {
        const int d = t + i * 256;
        x_acc[xo + d] += oproj[so + d] - nxr[so + d];
    }
}

// =====================================================================
// util
// =====================================================================
__global__ void cvt_bf16_kernel(const float* __restrict__ src, __bf16* __restrict__ dst, long n)
{
    const long i = (long)blockIdx.x * blockDim.x + threadIdx.x;
    if (i < n) dst[i] = (__bf16)src[i];
}

__global__ void zero_bf16_kernel(__bf16* __restrict__ p, long n)
{
    const long i = (long)blockIdx.x * blockDim.x + threadIdx.x;
    if (i < n) p[i] = (__bf16)0.0f;
}

// =====================================================================
// host
// =====================================================================
extern "C" void kernel_launch(void* const* d_in, const int* in_sizes, int n_in,
                              void* d_out, int out_size, void* d_ws, size_t ws_size,
                              hipStream_t stream)
{
    (void)in_sizes; (void)n_in; (void)out_size; (void)ws_size;
    const float* x    = (const float*)d_in[0];
    const float* wq_r = (const float*)d_in[1];
    const float* wk_r = (const float*)d_in[2];
    const float* Wq   = (const float*)d_in[3];
    const float* bq   = (const float*)d_in[4];
    const float* Wk   = (const float*)d_in[5];
    const float* bk   = (const float*)d_in[6];
    const float* Wv   = (const float*)d_in[7];
    const float* bv   = (const float*)d_in[8];
    const float* Wo   = (const float*)d_in[9];
    const float* bo   = (const float*)d_in[10];
    const float* W1   = (const float*)d_in[11];
    const float* b1   = (const float*)d_in[12];
    const float* W2   = (const float*)d_in[13];
    const float* b2   = (const float*)d_in[14];
    const float* g1   = (const float*)d_in[15];
    const float* be1  = (const float*)d_in[16];
    const float* g2   = (const float*)d_in[17];
    const float* be2  = (const float*)d_in[18];
    float* out = (float*)d_out;

    char* wp = (char*)d_ws;
    auto alloc = [&](size_t bytes) { char* p = wp; wp += (bytes + 255) & ~(size_t)255; return p; };

    const long BND = (long)B_ * N_ * D_;     // 8 Mi
    const long BKD = (long)B_ * KP_ * D_;
    const long DD  = (long)D_ * D_;
    const long D4D = (long)D_ * D4_;

    float*  x_acc = (float*) alloc(BND * 4);
    __bf16* n2b   = (__bf16*)alloc(BND * 2);
    __bf16* hb    = (__bf16*)alloc((long)B_ * N_ * D4_ * 2);
    __bf16* xsb   = (__bf16*)alloc(BKD * 2);
    float*  nxr   = (float*) alloc(BKD * 4);
    float*  qb    = (float*) alloc(BKD * 4);
    float*  kb    = (float*) alloc(BKD * 4);
    float*  vb    = (float*) alloc(BKD * 4);
    __bf16* ob    = (__bf16*)alloc(BKD * 2);
    float*  oproj = (float*) alloc(BKD * 4);
    float*  xbar  = (float*) alloc((long)B_ * D_ * 4);
    float*  qg    = (float*) alloc((long)B_ * R_ * 4);
    float*  wvecb = (float*) alloc((long)B_ * D_ * 4);
    float*  scb   = (float*) alloc((long)B_ * N_ * 4);
    int*    idxb  = (int*)   alloc((long)B_ * KP_ * 4);
    int*    posb  = (int*)   alloc((long)B_ * N_ * 4);
    __bf16* wqb   = (__bf16*)alloc(DD * 2);
    __bf16* wkb   = (__bf16*)alloc(DD * 2);
    __bf16* wvb   = (__bf16*)alloc(DD * 2);
    __bf16* wob   = (__bf16*)alloc(DD * 2);
    __bf16* w1b   = (__bf16*)alloc(D4D * 2);
    __bf16* w2b   = (__bf16*)alloc(D4D * 2);

    // ---- init ----
    (void)hipMemcpyAsync(x_acc, x, BND * 4, hipMemcpyDeviceToDevice, stream);
    zero_bf16_kernel<<<(unsigned)((BKD + 255) / 256), 256, 0, stream>>>(xsb, BKD);

    // ---- router ----
    colmean_kernel<<<(B_ * D_ + 255) / 256, 256, 0, stream>>>(x, xbar);
    qglobal_kernel<<<1, 64, 0, stream>>>(xbar, wq_r, qg);
    wvec_kernel<<<(B_ * D_ + 255) / 256, 256, 0, stream>>>(wk_r, qg, wvecb);
    scores_kernel<<<B_ * N_, 256, 0, stream>>>(x, wvecb, scb);
    topk_kernel<<<B_, 256, 0, stream>>>(scb, idxb, posb);

    const int MS = B_ * KP_;                         // 832 routed rows (padded)
    dim3 gQ(D_ / 128, (MS + 127) / 128);             // (8, 7)
    dim3 gF1(D4_ / 128, (B_ * N_) / 128);            // (32, 64)
    dim3 gF2(D_ / 128, (B_ * N_) / 128);             // (8, 64)

    for (int i = 0; i < L_; ++i) {
        // weight conversion (fp32 -> bf16), per layer, reused buffers
        cvt_bf16_kernel<<<(unsigned)((DD + 255) / 256), 256, 0, stream>>>(Wq + (long)i * DD, wqb, DD);
        cvt_bf16_kernel<<<(unsigned)((DD + 255) / 256), 256, 0, stream>>>(Wk + (long)i * DD, wkb, DD);
        cvt_bf16_kernel<<<(unsigned)((DD + 255) / 256), 256, 0, stream>>>(Wv + (long)i * DD, wvb, DD);
        cvt_bf16_kernel<<<(unsigned)((DD + 255) / 256), 256, 0, stream>>>(Wo + (long)i * DD, wob, DD);
        cvt_bf16_kernel<<<(unsigned)((D4D + 255) / 256), 256, 0, stream>>>(W1 + (long)i * D4D, w1b, D4D);
        cvt_bf16_kernel<<<(unsigned)((D4D + 255) / 256), 256, 0, stream>>>(W2 + (long)i * D4D, w2b, D4D);

        // LN1 + gather + residual-with-nx
        ln1_kernel<<<B_ * N_, 256, 0, stream>>>(x_acc, g1 + (long)i * D_, be1 + (long)i * D_,
                                                posb, xsb, nxr);

        // QKV projections (WMMA)
        gemm_wmma<0, 0, 0><<<gQ, 256, 0, stream>>>(xsb, wqb, bq + (long)i * D_, qb, nullptr, MS, D_, D_);
        gemm_wmma<0, 0, 0><<<gQ, 256, 0, stream>>>(xsb, wkb, bk + (long)i * D_, kb, nullptr, MS, D_, D_);
        gemm_wmma<0, 0, 0><<<gQ, 256, 0, stream>>>(xsb, wvb, bv + (long)i * D_, vb, nullptr, MS, D_, D_);

        // attention
        attn_kernel<<<B_ * H_, 256, 0, stream>>>(qb, kb, vb, ob);

        // output projection (WMMA)
        gemm_wmma<0, 0, 0><<<gQ, 256, 0, stream>>>(ob, wob, bo + (long)i * D_, oproj, nullptr, MS, D_, D_);

        // scatter correction at routed rows
        scatter_kernel<<<B_ * KT_, 256, 0, stream>>>(x_acc, oproj, nxr, idxb);

        // LN2 -> bf16
        ln2_kernel<<<B_ * N_, 256, 0, stream>>>(x_acc, g2 + (long)i * D_, be2 + (long)i * D_, n2b);

        // FFN1: gelu(n2 @ W1 + b1) -> bf16   (WMMA, dominant GEMM)
        gemm_wmma<1, 1, 0><<<gF1, 256, 0, stream>>>(n2b, w1b, b1 + (long)i * D4_, nullptr, hb,
                                                    B_ * N_, D4_, D_);
        // FFN2: x += h @ W2 + b2             (WMMA, accumulate into x)
        gemm_wmma<0, 0, 1><<<gF2, 256, 0, stream>>>(hb, w2b, b2 + (long)i * D_, x_acc, nullptr,
                                                    B_ * N_, D_, D4_);
    }

    (void)hipMemcpyAsync(out, x_acc, BND * 4, hipMemcpyDeviceToDevice, stream);
}
